// CrAKNAttention_1322849927202
// MI455X (gfx1250) — matrix operand order repaired and checked by reading, not compiled, not run on hardware
//
#include <hip/hip_runtime.h>
#include <hip/hip_bf16.h>
#include <math.h>

// Problem dims (fixed by the reference)
#define N_TOK 3072
#define D_IN  512
#define NH    16
#define HDIM  64
#define HD    1024   // NH*HDIM

typedef __bf16 bf16_t;
typedef __attribute__((ext_vector_type(16))) bf16_t v16bf;
typedef __attribute__((ext_vector_type(8)))  bf16_t v8bf;
typedef __attribute__((ext_vector_type(8)))  float  v8f;
typedef __attribute__((ext_vector_type(4)))  int    v4i;

#define AS1 __attribute__((address_space(1)))
#define AS3 __attribute__((address_space(3)))

#if defined(__has_builtin)
#if __has_builtin(__builtin_amdgcn_global_load_async_to_lds_b128)
#define HAVE_ASYNC_LDS 1
#endif
#if __has_builtin(__builtin_amdgcn_s_wait_asynccnt)
#define HAVE_WAIT_ASYNC 1
#endif
#endif

static __device__ __forceinline__ void async_copy16(const void* g, void* l) {
#ifdef HAVE_ASYNC_LDS
  // DMA 16 bytes global -> LDS without touching VGPRs; tracked by ASYNCcnt.
  __builtin_amdgcn_global_load_async_to_lds_b128((AS1 v4i*)g, (AS3 v4i*)l, 0, 0);
#else
  *(uint4*)l = *(const uint4*)g;   // fallback: bounce through VGPRs (DScnt)
#endif
}

#ifdef HAVE_WAIT_ASYNC
#define WAIT_ASYNC(n) __builtin_amdgcn_s_wait_asynccnt(n)
#else
#ifdef HAVE_ASYNC_LDS
#define WAIT_ASYNC(n) asm volatile("s_wait_asynccnt %0" ::"i"(n) : "memory")
#else
#define WAIT_ASYNC(n) ((void)0)
#endif
#endif

static __device__ __forceinline__ bf16_t f2bf(float f) {
  unsigned u = __builtin_bit_cast(unsigned, f);
  unsigned r = u + 0x7FFFu + ((u >> 16) & 1u);   // round-to-nearest-even
  unsigned short h = (unsigned short)(r >> 16);
  return __builtin_bit_cast(bf16_t, h);
}

static __device__ __forceinline__ v16bf cat8(v8bf lo, v8bf hi) {
  return __builtin_shufflevector(lo, hi, 0,1,2,3,4,5,6,7,8,9,10,11,12,13,14,15);
}

static __device__ __forceinline__ v8f wmma_bf16(v16bf a, v16bf b, v8f c) {
  // D(f32 16x16) = A(16x32 bf16) * B(32x16 bf16) + C
  return __builtin_amdgcn_wmma_f32_16x16x32_bf16(false, a, false, b, (short)0, c,
                                                 false, false);
}

static __device__ __forceinline__ float mishf(float x) {
  float sp = (x > 20.f) ? x : log1pf(__expf(x));
  return x * tanhf(sp);
}

// ---------------------------------------------------------------- convert
__global__ void to_bf16_kernel(const float* __restrict__ src,
                               bf16_t* __restrict__ dst, int n) {
  int i = blockIdx.x * blockDim.x + threadIdx.x;
  if (i < n) dst[i] = f2bf(src[i]);
}

// ---------------------------------------------------------------- projections
// out = X @ W^T + b.  X:[rows,K] bf16, W:[cols,K] bf16 (both row-major).
// One wave -> 16x64 output strip (4 WMMA C tiles), A fragment reused 4x.
// vmode==0: out[h][n][d] (Q,K head-major);  vmode==1: out[h][d][n] (V^T).
__global__ __launch_bounds__(256) void proj_kernel(
    const bf16_t* __restrict__ X, const bf16_t* __restrict__ W,
    const float* __restrict__ bvec, bf16_t* __restrict__ out, int vmode) {
  const int lane = threadIdx.x & 31;
  const int wave = threadIdx.x >> 5;
  const int half = lane >> 4;
  const int l15  = lane & 15;
  const int tile = blockIdx.x * 8 + wave;
  const int colGroups = HD / 64;
  const int n0 = (tile / colGroups) * 16;
  const int j0 = (tile % colGroups) * 64;

  const bf16_t* arow = X + (n0 + l15) * D_IN + half * 16;
  const bf16_t* b0 = W + (j0 + l15) * D_IN + half * 16;
  const bf16_t* b1 = b0 + 16 * D_IN;
  const bf16_t* b2 = b0 + 32 * D_IN;
  const bf16_t* b3 = b0 + 48 * D_IN;

  v8f c0 = {}, c1 = {}, c2 = {}, c3 = {};
  #pragma unroll 2
  for (int k0 = 0; k0 < D_IN; k0 += 32) {
    v16bf a = cat8(*(const v8bf*)(arow + k0), *(const v8bf*)(arow + k0 + 8));
    c0 = wmma_bf16(a, *(const v16bf*)(b0 + k0), c0);
    c1 = wmma_bf16(a, *(const v16bf*)(b1 + k0), c1);
    c2 = wmma_bf16(a, *(const v16bf*)(b2 + k0), c2);
    c3 = wmma_bf16(a, *(const v16bf*)(b3 + k0), c3);
  }
  #pragma unroll
  for (int t = 0; t < 4; ++t) {
    const v8f c = (t == 0) ? c0 : (t == 1) ? c1 : (t == 2) ? c2 : c3;
    const int j = j0 + t * 16 + l15;
    const float bb = bvec[j];
    const int h = j >> 6, d = j & 63;
    #pragma unroll
    for (int r = 0; r < 8; ++r) {
      const int m = n0 + r + 8 * half;     // C layout: VGPR r -> row r+8*half
      const float v = c[r] + bb;
      if (vmode == 0) out[((h * N_TOK) + m) * HDIM + d] = f2bf(v);
      else            out[((h * HDIM) + d) * N_TOK + m] = f2bf(v);
    }
  }
}

// ---------------------------------------------------------------- fused attention
// Flash-style. 8 waves/block all belong to ONE head (192 row-tiles % 8 == 0),
// so the block cooperatively double-buffers each 32-key K/V chunk in LDS via
// async global->LDS DMA; waves consume it with ds reads feeding WMMA.
__global__ __launch_bounds__(256) void attn_kernel(
    const bf16_t* __restrict__ qh, const bf16_t* __restrict__ kh,
    const bf16_t* __restrict__ vT, const float* __restrict__ bias,
    const int* __restrict__ mask, bf16_t* __restrict__ vals) {
  __shared__ bf16_t Kl[2][32][64];     // [buf][key][d]      4KB each
  __shared__ bf16_t Vl[2][64][32];     // [buf][d][key]      4KB each
  __shared__ bf16_t Pl[8][16][40];     // wave-private P tile (+pad)
  const int tid  = threadIdx.x;
  const int lane = tid & 31;
  const int wave = tid >> 5;
  const int half = lane >> 4;
  const int l15  = lane & 15;
  const int h  = blockIdx.x / 24;                       // 24 blocks per head
  const int n0 = ((blockIdx.x % 24) * 8 + wave) * 16;

  const bf16_t* khBase = kh + (size_t)h * N_TOK * HDIM; // [n][d]
  const bf16_t* vTBase = vT + (size_t)h * HDIM * N_TOK; // [d][n]

  // Preload Q A-tiles for the full K=64 reduction, kept in regs.
  const bf16_t* qrow = qh + ((h * N_TOK) + n0 + l15) * HDIM + half * 16;
  const v16bf aq0 = cat8(*(const v8bf*)(qrow),      *(const v8bf*)(qrow + 8));
  const v16bf aq1 = cat8(*(const v8bf*)(qrow + 32), *(const v8bf*)(qrow + 40));

  v8f acc0 = {}, acc1 = {}, acc2 = {}, acc3 = {};
  float mI[8], lI[8];
  #pragma unroll
  for (int r = 0; r < 8; ++r) { mI[r] = -INFINITY; lI[r] = 0.f; }

  // Cooperative stage of one 32-key chunk: K chunk is 4KB contiguous, V chunk
  // is 64 rows x 64B. 256 threads x 16B each per array.
  auto stage = [&](int buf, int j) {
    async_copy16(khBase + (size_t)j * HDIM + tid * 8, &Kl[buf][0][0] + tid * 8);
    const int d = tid >> 2, seg = tid & 3;
    async_copy16(vTBase + (size_t)d * N_TOK + j + seg * 8, &Vl[buf][d][seg * 8]);
  };

  const float scale = 0.125f;  // 1/sqrt(64)
  stage(0, 0);
  for (int i = 0; i < N_TOK / 32; ++i) {
    const int j = i * 32;
    const int cur = i & 1;
    if (i + 1 < N_TOK / 32) { stage(cur ^ 1, j + 32); WAIT_ASYNC(2); }
    else                    { WAIT_ASYNC(0); }
    __syncthreads();                       // chunk `cur` visible to all waves

    // ---- S = Q K^T (two 16-col tiles, K=64 over two WMMA steps), from LDS
    v8f s0 = {}, s1 = {};
    s0 = wmma_bf16(aq0, *(const v16bf*)&Kl[cur][l15][half * 16],           s0);
    s0 = wmma_bf16(aq1, *(const v16bf*)&Kl[cur][l15][32 + half * 16],      s0);
    s1 = wmma_bf16(aq0, *(const v16bf*)&Kl[cur][16 + l15][half * 16],      s1);
    s1 = wmma_bf16(aq1, *(const v16bf*)&Kl[cur][16 + l15][32 + half * 16], s1);

    // ---- scale + bias + mask (bias/mask stream from L2, reused across heads)
    float cmax[8];
    #pragma unroll
    for (int r = 0; r < 8; ++r) {
      const int m = n0 + r + 8 * half;
      const long long o0 = (long long)m * N_TOK + (j + l15);
      const float v0 = (mask[o0]      == 0) ? -9.0e15f : (s0[r] * scale + bias[o0]);
      const float v1 = (mask[o0 + 16] == 0) ? -9.0e15f : (s1[r] * scale + bias[o0 + 16]);
      s0[r] = v0; s1[r] = v1;
      cmax[r] = fmaxf(v0, v1);
    }
    // ---- online softmax (row reductions stay inside each 16-lane half)
    float p0[8], p1[8], alpha[8];
    #pragma unroll
    for (int r = 0; r < 8; ++r) {
      float x = cmax[r];
      x = fmaxf(x, __shfl_xor(x, 1));
      x = fmaxf(x, __shfl_xor(x, 2));
      x = fmaxf(x, __shfl_xor(x, 4));
      x = fmaxf(x, __shfl_xor(x, 8));
      const float mNew = fmaxf(mI[r], x);
      alpha[r] = __expf(mI[r] - mNew);
      mI[r] = mNew;
    }
    #pragma unroll
    for (int r = 0; r < 8; ++r) {
      p0[r] = __expf(s0[r] - mI[r]);
      p1[r] = __expf(s1[r] - mI[r]);
      float s = p0[r] + p1[r];
      s += __shfl_xor(s, 1);
      s += __shfl_xor(s, 2);
      s += __shfl_xor(s, 4);
      s += __shfl_xor(s, 8);
      lI[r] = lI[r] * alpha[r] + s;
      acc0[r] *= alpha[r]; acc1[r] *= alpha[r];
      acc2[r] *= alpha[r]; acc3[r] *= alpha[r];
    }
    // ---- C-layout -> A-layout conversion of P via wave-private LDS
    #pragma unroll
    for (int r = 0; r < 8; ++r) {
      const int row = r + 8 * half;
      Pl[wave][row][l15]      = f2bf(p0[r]);
      Pl[wave][row][16 + l15] = f2bf(p1[r]);
    }
    asm volatile("s_wait_dscnt 0" ::: "memory");
    const bf16_t* prow = &Pl[wave][l15][half * 16];
    const v16bf pa = cat8(*(const v8bf*)(prow), *(const v8bf*)(prow + 8));
    // ---- O += P @ V, V chunk from LDS ([d][key]: contiguous B columns)
    acc0 = wmma_bf16(pa, *(const v16bf*)&Vl[cur][l15][half * 16],      acc0);
    acc1 = wmma_bf16(pa, *(const v16bf*)&Vl[cur][16 + l15][half * 16], acc1);
    acc2 = wmma_bf16(pa, *(const v16bf*)&Vl[cur][32 + l15][half * 16], acc2);
    acc3 = wmma_bf16(pa, *(const v16bf*)&Vl[cur][48 + l15][half * 16], acc3);
    __syncthreads();                       // all waves done reading `cur`
  }
  // ---- normalize and store concatenated heads [N_TOK, HD]
  #pragma unroll
  for (int r = 0; r < 8; ++r) {
    const float inv = 1.0f / lI[r];
    const int m = n0 + r + 8 * half;
    bf16_t* o = vals + (long long)m * HD + h * HDIM + l15;
    o[0]  = f2bf(acc0[r] * inv);
    o[16] = f2bf(acc1[r] * inv);
    o[32] = f2bf(acc2[r] * inv);
    o[48] = f2bf(acc3[r] * inv);
  }
}

// ---------------------------------------------------------------- MLP
// One wave -> 16x64 output strip, K = HD.
__global__ __launch_bounds__(256) void mlp1_kernel(
    const bf16_t* __restrict__ X, const bf16_t* __restrict__ W,
    const float* __restrict__ bvec, bf16_t* __restrict__ out) {
  const int lane = threadIdx.x & 31;
  const int wave = threadIdx.x >> 5;
  const int half = lane >> 4;
  const int l15  = lane & 15;
  const int tile = blockIdx.x * 8 + wave;
  const int colGroups = HD / 64;
  const int n0 = (tile / colGroups) * 16;
  const int j0 = (tile % colGroups) * 64;

  const bf16_t* arow = X + (n0 + l15) * HD + half * 16;
  const bf16_t* b0 = W + (j0 + l15) * HD + half * 16;
  const bf16_t* b1 = b0 + 16 * HD;
  const bf16_t* b2 = b0 + 32 * HD;
  const bf16_t* b3 = b0 + 48 * HD;
  v8f c0 = {}, c1 = {}, c2 = {}, c3 = {};
  #pragma unroll 2
  for (int k0 = 0; k0 < HD; k0 += 32) {
    v16bf a = cat8(*(const v8bf*)(arow + k0), *(const v8bf*)(arow + k0 + 8));
    c0 = wmma_bf16(a, *(const v16bf*)(b0 + k0), c0);
    c1 = wmma_bf16(a, *(const v16bf*)(b1 + k0), c1);
    c2 = wmma_bf16(a, *(const v16bf*)(b2 + k0), c2);
    c3 = wmma_bf16(a, *(const v16bf*)(b3 + k0), c3);
  }
  #pragma unroll
  for (int t = 0; t < 4; ++t) {
    const v8f c = (t == 0) ? c0 : (t == 1) ? c1 : (t == 2) ? c2 : c3;
    const int j = j0 + t * 16 + l15;
    const float bb = bvec[j];
    #pragma unroll
    for (int r = 0; r < 8; ++r) {
      const int m = n0 + r + 8 * half;
      out[m * HD + j] = f2bf(mishf(c[r] + bb));
    }
  }
}

__global__ __launch_bounds__(256) void mlp2_kernel(
    const bf16_t* __restrict__ X, const bf16_t* __restrict__ W,
    const float* __restrict__ bvec, float* __restrict__ out) {
  const int lane = threadIdx.x & 31;
  const int wave = threadIdx.x >> 5;
  const int half = lane >> 4;
  const int l15  = lane & 15;
  const int tile = blockIdx.x * 8 + wave;
  const int colGroups = D_IN / 64;
  const int n0 = (tile / colGroups) * 16;
  const int j0 = (tile % colGroups) * 64;

  const bf16_t* arow = X + (n0 + l15) * HD + half * 16;
  const bf16_t* b0 = W + (j0 + l15) * HD + half * 16;
  const bf16_t* b1 = b0 + 16 * HD;
  const bf16_t* b2 = b0 + 32 * HD;
  const bf16_t* b3 = b0 + 48 * HD;
  v8f c0 = {}, c1 = {}, c2 = {}, c3 = {};
  #pragma unroll 2
  for (int k0 = 0; k0 < HD; k0 += 32) {
    v16bf a = cat8(*(const v8bf*)(arow + k0), *(const v8bf*)(arow + k0 + 8));
    c0 = wmma_bf16(a, *(const v16bf*)(b0 + k0), c0);
    c1 = wmma_bf16(a, *(const v16bf*)(b1 + k0), c1);
    c2 = wmma_bf16(a, *(const v16bf*)(b2 + k0), c2);
    c3 = wmma_bf16(a, *(const v16bf*)(b3 + k0), c3);
  }
  #pragma unroll
  for (int t = 0; t < 4; ++t) {
    const v8f c = (t == 0) ? c0 : (t == 1) ? c1 : (t == 2) ? c2 : c3;
    const int j = j0 + t * 16 + l15;
    const float bb = bvec[j];
    #pragma unroll
    for (int r = 0; r < 8; ++r) {
      const int m = n0 + r + 8 * half;
      out[m * D_IN + j] = c[r] + bb;
    }
  }
}

// ---------------------------------------------------------------- launch
extern "C" void kernel_launch(void* const* d_in, const int* in_sizes, int n_in,
                              void* d_out, int out_size, void* d_ws, size_t ws_size,
                              hipStream_t stream) {
  const float* q    = (const float*)d_in[0];
  const float* k    = (const float*)d_in[1];
  const float* v    = (const float*)d_in[2];
  const float* bias = (const float*)d_in[3];
  const int*   mask = (const int*)d_in[4];
  const float* Wq   = (const float*)d_in[5];
  const float* bq   = (const float*)d_in[6];
  const float* Wk   = (const float*)d_in[7];
  const float* bk   = (const float*)d_in[8];
  const float* Wv   = (const float*)d_in[9];
  const float* bv   = (const float*)d_in[10];
  const float* Wo1  = (const float*)d_in[11];
  const float* bo1  = (const float*)d_in[12];
  const float* Wo2  = (const float*)d_in[13];
  const float* bo2  = (const float*)d_in[14];

  // workspace carve-up (~47 MB total)
  char* ws = (char*)d_ws;
  size_t off = 0;
  auto carve = [&](size_t bytes) {
    char* p = ws + off;
    off += (bytes + 255) & ~size_t(255);
    return p;
  };
  bf16_t* qb   = (bf16_t*)carve((size_t)N_TOK * D_IN * 2);
  bf16_t* kb   = (bf16_t*)carve((size_t)N_TOK * D_IN * 2);
  bf16_t* vb   = (bf16_t*)carve((size_t)N_TOK * D_IN * 2);
  bf16_t* Wqb  = (bf16_t*)carve((size_t)HD * D_IN * 2);
  bf16_t* Wkb  = (bf16_t*)carve((size_t)HD * D_IN * 2);
  bf16_t* Wvb  = (bf16_t*)carve((size_t)HD * D_IN * 2);
  bf16_t* Wo1b = (bf16_t*)carve((size_t)HD * HD * 2);
  bf16_t* Wo2b = (bf16_t*)carve((size_t)D_IN * HD * 2);
  bf16_t* qhh  = (bf16_t*)carve((size_t)NH * N_TOK * HDIM * 2);  // [H][N][D]
  bf16_t* khh  = (bf16_t*)carve((size_t)NH * N_TOK * HDIM * 2);  // [H][N][D]
  bf16_t* vTT  = (bf16_t*)carve((size_t)NH * HDIM * N_TOK * 2);  // [H][D][N]
  bf16_t* vals = (bf16_t*)carve((size_t)N_TOK * HD * 2);
  bf16_t* hid  = (bf16_t*)carve((size_t)N_TOK * HD * 2);

  auto cvt = [&](const float* s, bf16_t* d, int n) {
    to_bf16_kernel<<<(n + 255) / 256, 256, 0, stream>>>(s, d, n);
  };
  cvt(q, qb, N_TOK * D_IN);
  cvt(k, kb, N_TOK * D_IN);
  cvt(v, vb, N_TOK * D_IN);
  cvt(Wq, Wqb, HD * D_IN);
  cvt(Wk, Wkb, HD * D_IN);
  cvt(Wv, Wvb, HD * D_IN);
  cvt(Wo1, Wo1b, HD * HD);
  cvt(Wo2, Wo2b, D_IN * HD);

  const int projBlocks = (N_TOK / 16) * (HD / 64) / 8;   // 384
  proj_kernel<<<projBlocks, 256, 0, stream>>>(qb, Wqb, bq, qhh, 0);
  proj_kernel<<<projBlocks, 256, 0, stream>>>(kb, Wkb, bk, khh, 0);
  proj_kernel<<<projBlocks, 256, 0, stream>>>(vb, Wvb, bv, vTT, 1);

  const int attnBlocks = NH * (N_TOK / 16) / 8;          // 384
  attn_kernel<<<attnBlocks, 256, 0, stream>>>(qhh, khh, vTT, bias, mask, vals);

  mlp1_kernel<<<projBlocks, 256, 0, stream>>>(vals, Wo1b, bo1, hid);
  const int mlp2Blocks = (N_TOK / 16) * (D_IN / 64) / 8; // 192
  mlp2_kernel<<<mlp2Blocks, 256, 0, stream>>>(hid, Wo2b, bo2, (float*)d_out);

  // second tuple output: bias passthrough
  (void)hipMemcpyAsync((char*)d_out + (size_t)N_TOK * D_IN * sizeof(float), bias,
                       (size_t)N_TOK * N_TOK * sizeof(float),
                       hipMemcpyDeviceToDevice, stream);
}
